// AP_19258633355825
// MI455X (gfx1250) — compile-verified, matrix operand order checked by reading, not run on hardware
//
#include <hip/hip_runtime.h>
#include <stdint.h>

#define BATCH  256
#define NPROP  4000
#define NGT    50
#define MTOT   (BATCH*NPROP)       /* 1,024,000 */
#define NBUCK  65536
#define CHUNK  4096
#define NCHUNK (MTOT/CHUNK)        /* 250 */
#define PWORDS 128                 /* 125 used words, padded to 128 */
#define NLABELS 12800.0f

typedef __attribute__((ext_vector_type(16))) _Float16 v16h;
typedef __attribute__((ext_vector_type(8)))  float    v8f;

__device__ __forceinline__ int bucket_of(float c) {
    int b = (int)(c * 65536.0f);
    return min(max(b, 0), NBUCK - 1);
}

// ---------------------------------------------------------------------------
// K1: per-batch greedy TP matching (both thresholds). One block per batch.
// pot bitmasks built with wave32 __ballot; greedy argmin over 128 LDS words.
// ---------------------------------------------------------------------------
__global__ __launch_bounds__(256) void k_tp(const float* __restrict__ segments,
                                            const float* __restrict__ labels,
                                            unsigned char* __restrict__ flags)
{
    __shared__ float    gt[NGT * 2];
    __shared__ unsigned pot[2][NGT][PWORDS];   // 51,200 B
    __shared__ unsigned used[2][PWORDS];       //  1,024 B
    __shared__ int      candIdx[PWORDS];

    const int b    = blockIdx.x;
    const int tid  = threadIdx.x;
    const int lane = tid & 31;

    unsigned* pz = &pot[0][0][0];
    for (int i = tid; i < 2 * NGT * PWORDS; i += 256) pz[i] = 0;
    ((unsigned*)used)[tid] = 0;                        // exactly 256 words
    if (tid < NGT * 2) gt[tid] = labels[b * NGT * 2 + tid];
    __syncthreads();

    const float2* seg2 = (const float2*)segments;
    for (int i = tid; i < NPROP; i += 256) {           // uniform per wave (4000 = 125*32)
        float2 s = seg2[b * NPROP + i];
        __builtin_prefetch(&seg2[b * NPROP + i + 256], 0, 0);
        const int w = i >> 5;
        #pragma unroll 1
        for (int g = 0; g < NGT; ++g) {
            float bmin = gt[g * 2 + 0], bmax = gt[g * 2 + 1];
            float inter = fminf(s.y, bmax) - fmaxf(s.x, bmin);
            inter = fmaxf(inter, 0.0f);
            float uni = (s.y - s.x) + (bmax - bmin) - inter;
            float iou = inter / uni;
            unsigned m0 = (unsigned)__ballot(iou > 0.50f);
            unsigned m1 = (unsigned)__ballot(iou > 0.75f);
            if (lane == 0) { pot[0][g][w] = m0; pot[1][g][w] = m1; }
        }
    }
    __syncthreads();

    for (int t = 0; t < 2; ++t) {
        for (int g = 0; g < NGT; ++g) {
            if (tid < PWORDS) {
                unsigned c = pot[t][g][tid] & ~used[t][tid];
                candIdx[tid] = c ? (tid * 32 + __ffs(c) - 1) : 0x7fffffff;
            }
            __syncthreads();
            if (tid == 0) {
                int mn = 0x7fffffff;
                for (int w2 = 0; w2 < PWORDS; ++w2) mn = min(mn, candIdx[w2]);
                if (mn != 0x7fffffff) used[t][mn >> 5] |= (1u << (mn & 31));
            }
            __syncthreads();
        }
    }

    for (int i = tid; i < NPROP; i += 256) {
        unsigned f = ((used[0][i >> 5] >> (i & 31)) & 1u)
                   | (((used[1][i >> 5] >> (i & 31)) & 1u) << 1);
        flags[b * NPROP + i] = (unsigned char)f;
    }
}

// ---------------------------------------------------------------------------
// K3: bucket histogram of confidences
// ---------------------------------------------------------------------------
__global__ void k_hist(const float* __restrict__ scores, unsigned* __restrict__ hist)
{
    int i = blockIdx.x * blockDim.x + threadIdx.x;
    if (i < MTOT) atomicAdd(&hist[bucket_of(scores[i])], 1u);
}

// ---------------------------------------------------------------------------
// K4: descending exclusive scan -> start[b] = #elements in buckets > b
// ---------------------------------------------------------------------------
__global__ __launch_bounds__(1024) void k_scan(const unsigned* __restrict__ hist,
                                               unsigned* __restrict__ start)
{
    __shared__ unsigned tot[1024];
    const int t = threadIdx.x;
    unsigned s = 0;
    for (int j = 0; j < 64; ++j) s += hist[t * 64 + j];
    tot[t] = s;
    __syncthreads();
    unsigned suf = 0;
    for (int t2 = t + 1; t2 < 1024; ++t2) suf += tot[t2];
    unsigned run = suf;
    for (int j = 63; j >= 0; --j) {
        int bb = t * 64 + j;
        start[bb] = run;
        run += hist[bb];
    }
}

// ---------------------------------------------------------------------------
// K5: scatter (key = conf bits, val = flags<<24 | origIdx) into bucket ranges
// ---------------------------------------------------------------------------
__global__ void k_scatter(const float* __restrict__ scores,
                          const unsigned char* __restrict__ flags,
                          const unsigned* __restrict__ start,
                          unsigned* __restrict__ cursor,
                          unsigned* __restrict__ keys,
                          unsigned* __restrict__ vals)
{
    int i = blockIdx.x * blockDim.x + threadIdx.x;
    if (i >= MTOT) return;
    float c = scores[i];
    int b = bucket_of(c);
    unsigned pos = start[b] + atomicAdd(&cursor[b], 1u);
    keys[pos] = __float_as_uint(c);        // conf >= 0: uint order == float order
    vals[pos] = ((unsigned)flags[i] << 24) | (unsigned)i;   // MTOT < 2^24
}

// ---------------------------------------------------------------------------
// K6: exact stable rank within each bucket (desc conf, ties: orig idx asc);
// emit tp flag byte to its exact global sorted slot. One thread per bucket.
// ---------------------------------------------------------------------------
__global__ void k_rank(const unsigned* __restrict__ hist,
                       const unsigned* __restrict__ start,
                       const unsigned* __restrict__ keys,
                       const unsigned* __restrict__ vals,
                       unsigned char* __restrict__ tps)
{
    int b = blockIdx.x * blockDim.x + threadIdx.x;
    if (b >= NBUCK) return;
    const unsigned s = start[b], c = hist[b];
    for (unsigned e = 0; e < c; ++e) {
        unsigned k   = keys[s + e];
        unsigned v   = vals[s + e];
        unsigned idx = v & 0x00FFFFFFu;
        unsigned r = 0;
        for (unsigned e2 = 0; e2 < c; ++e2) {
            unsigned k2 = keys[s + e2];
            if (k2 > k) { r++; }
            else if (k2 == k) {
                if ((vals[s + e2] & 0x00FFFFFFu) < idx) r++;
            }
        }
        tps[s + r] = (unsigned char)(v >> 24);
    }
}

// ---------------------------------------------------------------------------
// K7: per-chunk TP sums via WMMA. A = tp bits (f16 0/1), B = ones, C = 0.
// sum(D) = 16 * sum(A): layout-insensitive, numerically exact.
// ---------------------------------------------------------------------------
__global__ __launch_bounds__(256) void k_chunksum(const unsigned char* __restrict__ tps,
                                                  float* __restrict__ chunkSums)
{
    const int c = blockIdx.x, tid = threadIdx.x;
    const int wave = tid >> 5, lane = tid & 31;
    const unsigned char* p = tps + (size_t)c * CHUNK + wave * 512 + lane * 16;
    __builtin_prefetch(p + 4096, 0, 0);

    v16h a0, a1, ones;
    #pragma unroll
    for (int j = 0; j < 16; ++j) {
        unsigned f = p[j];
        a0[j]  = (_Float16)(float)(f & 1u);
        a1[j]  = (_Float16)(float)((f >> 1) & 1u);
        ones[j] = (_Float16)1.0f;
    }
    v8f zc = {0.f, 0.f, 0.f, 0.f, 0.f, 0.f, 0.f, 0.f};
    v8f d0 = __builtin_amdgcn_wmma_f32_16x16x32_f16(false, a0, false, ones,
                                                    (short)0, zc, false, false);
    v8f d1 = __builtin_amdgcn_wmma_f32_16x16x32_f16(false, a1, false, ones,
                                                    (short)0, zc, false, false);
    float s0 = 0.f, s1 = 0.f;
    #pragma unroll
    for (int j = 0; j < 8; ++j) { s0 += d0[j]; s1 += d1[j]; }
    #pragma unroll
    for (int o = 16; o >= 1; o >>= 1) {
        s0 += __shfl_xor(s0, o, 32);
        s1 += __shfl_xor(s1, o, 32);
    }
    s0 *= (1.0f / 16.0f);
    s1 *= (1.0f / 16.0f);

    __shared__ float wsum[8][2];
    if (lane == 0) { wsum[wave][0] = s0; wsum[wave][1] = s1; }
    __syncthreads();
    if (tid == 0) {
        float t0 = 0.f, t1 = 0.f;
        for (int w = 0; w < 8; ++w) { t0 += wsum[w][0]; t1 += wsum[w][1]; }
        chunkSums[c * 2 + 0] = t0;
        chunkSums[c * 2 + 1] = t1;
    }
}

// K8: forward exclusive scan of chunk sums (serial, deterministic)
__global__ void k_basescan(const float* __restrict__ chunkSums, float* __restrict__ chunkBase)
{
    if (threadIdx.x == 0 && blockIdx.x == 0) {
        float a0 = 0.f, a1 = 0.f;
        for (int c = 0; c < NCHUNK; ++c) {
            chunkBase[c * 2 + 0] = a0;
            chunkBase[c * 2 + 1] = a1;
            a0 += chunkSums[c * 2 + 0];
            a1 += chunkSums[c * 2 + 1];
        }
    }
}

// ---------------------------------------------------------------------------
// K9: per-chunk max precision (needs intra-chunk prefix of tp)
// ---------------------------------------------------------------------------
__global__ __launch_bounds__(256) void k_maxp(const unsigned char* __restrict__ tps,
                                              const float* __restrict__ chunkBase,
                                              float* __restrict__ maxP)
{
    __shared__ int   tot0[256], tot1[256];
    __shared__ float mx0[256], mx1[256];
    const int c = blockIdx.x, t = threadIdx.x;
    const unsigned char* p = tps + (size_t)c * CHUNK + t * 16;

    int f[16], pre0[16], pre1[16];
    int s0 = 0, s1 = 0;
    #pragma unroll
    for (int j = 0; j < 16; ++j) {
        f[j] = p[j];
        s0 += f[j] & 1;        pre0[j] = s0;
        s1 += (f[j] >> 1) & 1; pre1[j] = s1;
    }
    tot0[t] = s0; tot1[t] = s1;
    __syncthreads();
    int e0 = 0, e1 = 0;
    for (int t2 = 0; t2 < t; ++t2) { e0 += tot0[t2]; e1 += tot1[t2]; }
    const float b0 = chunkBase[c * 2 + 0], b1 = chunkBase[c * 2 + 1];
    float m0 = -1.0f, m1 = -1.0f;
    #pragma unroll
    for (int j = 0; j < 16; ++j) {
        const int kg = c * CHUNK + t * 16 + j;
        const float denom = (float)(kg + 1);
        m0 = fmaxf(m0, (b0 + (float)(e0 + pre0[j])) / denom);
        m1 = fmaxf(m1, (b1 + (float)(e1 + pre1[j])) / denom);
    }
    mx0[t] = m0; mx1[t] = m1;
    __syncthreads();
    if (t == 0) {
        float a = -1.0f, b = -1.0f;
        for (int t2 = 0; t2 < 256; ++t2) { a = fmaxf(a, mx0[t2]); b = fmaxf(b, mx1[t2]); }
        maxP[c * 2 + 0] = a;
        maxP[c * 2 + 1] = b;
    }
}

// K10: suffix max over chunks (exclusive: max over strictly-later chunks)
__global__ void k_sufmax(const float* __restrict__ maxP, float* __restrict__ sufAfter)
{
    if (threadIdx.x == 0 && blockIdx.x == 0) {
        float r0 = -1.0f, r1 = -1.0f;
        for (int c = NCHUNK - 1; c >= 0; --c) {
            sufAfter[c * 2 + 0] = r0;
            sufAfter[c * 2 + 1] = r1;
            r0 = fmaxf(r0, maxP[c * 2 + 0]);
            r1 = fmaxf(r1, maxP[c * 2 + 1]);
        }
    }
}

// ---------------------------------------------------------------------------
// K11: per-chunk sum of tp_k * M_k (M_k = inclusive suffix-max of precision),
// excluding global k == 0. Fixed-order reductions for determinism.
// ---------------------------------------------------------------------------
__global__ __launch_bounds__(256) void k_dot(const unsigned char* __restrict__ tps,
                                             const float* __restrict__ chunkBase,
                                             const float* __restrict__ sufAfter,
                                             float* __restrict__ chunkDot)
{
    __shared__ int   tot0[256], tot1[256];
    __shared__ float mx0[256], mx1[256];
    __shared__ float pt0[256], pt1[256];
    const int c = blockIdx.x, t = threadIdx.x;
    const unsigned char* p = tps + (size_t)c * CHUNK + t * 16;

    int f[16], pre0[16], pre1[16];
    int s0 = 0, s1 = 0;
    #pragma unroll
    for (int j = 0; j < 16; ++j) {
        f[j] = p[j];
        s0 += f[j] & 1;        pre0[j] = s0;
        s1 += (f[j] >> 1) & 1; pre1[j] = s1;
    }
    tot0[t] = s0; tot1[t] = s1;
    __syncthreads();
    int e0 = 0, e1 = 0;
    for (int t2 = 0; t2 < t; ++t2) { e0 += tot0[t2]; e1 += tot1[t2]; }
    const float b0 = chunkBase[c * 2 + 0], b1 = chunkBase[c * 2 + 1];

    float pv0[16], pv1[16];
    float m0 = -1.0f, m1 = -1.0f;
    #pragma unroll
    for (int j = 0; j < 16; ++j) {
        const int kg = c * CHUNK + t * 16 + j;
        const float denom = (float)(kg + 1);
        pv0[j] = (b0 + (float)(e0 + pre0[j])) / denom;
        pv1[j] = (b1 + (float)(e1 + pre1[j])) / denom;
        m0 = fmaxf(m0, pv0[j]);
        m1 = fmaxf(m1, pv1[j]);
    }
    mx0[t] = m0; mx1[t] = m1;
    __syncthreads();
    float run0 = sufAfter[c * 2 + 0], run1 = sufAfter[c * 2 + 1];
    for (int t2 = t + 1; t2 < 256; ++t2) {
        run0 = fmaxf(run0, mx0[t2]);
        run1 = fmaxf(run1, mx1[t2]);
    }
    float term0 = 0.f, term1 = 0.f;
    #pragma unroll
    for (int j = 15; j >= 0; --j) {
        run0 = fmaxf(run0, pv0[j]);
        run1 = fmaxf(run1, pv1[j]);
        const int kg = c * CHUNK + t * 16 + j;
        if (kg >= 1) {
            if (f[j] & 1) term0 += run0;
            if (f[j] & 2) term1 += run1;
        }
    }
    pt0[t] = term0; pt1[t] = term1;
    __syncthreads();
    if (t == 0) {
        float a = 0.f, b = 0.f;
        for (int t2 = 0; t2 < 256; ++t2) { a += pt0[t2]; b += pt1[t2]; }
        chunkDot[c * 2 + 0] = a;
        chunkDot[c * 2 + 1] = b;
    }
}

// K12: final deterministic reduction -> AP[2]
__global__ void k_final(const float* __restrict__ chunkDot, float* __restrict__ out)
{
    if (threadIdx.x == 0 && blockIdx.x == 0) {
        float a0 = 0.f, a1 = 0.f;
        for (int c = 0; c < NCHUNK; ++c) { a0 += chunkDot[c * 2 + 0]; a1 += chunkDot[c * 2 + 1]; }
        out[0] = a0 / NLABELS;
        out[1] = a1 / NLABELS;
    }
}

// ---------------------------------------------------------------------------
extern "C" void kernel_launch(void* const* d_in, const int* in_sizes, int n_in,
                              void* d_out, int out_size, void* d_ws, size_t ws_size,
                              hipStream_t stream)
{
    (void)in_sizes; (void)n_in; (void)out_size; (void)ws_size;
    const float* scores   = (const float*)d_in[0];
    const float* segments = (const float*)d_in[1];
    const float* labels   = (const float*)d_in[2];
    float* out = (float*)d_out;

    char* ws = (char*)d_ws;
    size_t o = 0;
    auto alloc = [&](size_t bytes) { size_t r = o; o += (bytes + 255) & ~(size_t)255; return r; };

    size_t off_flags  = alloc(MTOT);            // u8
    size_t off_hist   = alloc(NBUCK * 4);       // u32
    size_t off_cursor = alloc(NBUCK * 4);       // u32 (adjacent to hist)
    size_t off_start  = alloc(NBUCK * 4);       // u32
    size_t off_keys   = alloc((size_t)MTOT * 4);
    size_t off_vals   = alloc((size_t)MTOT * 4);
    size_t off_tps    = alloc(MTOT);            // u8
    size_t off_csum   = alloc(NCHUNK * 2 * 4);
    size_t off_cbase  = alloc(NCHUNK * 2 * 4);
    size_t off_maxp   = alloc(NCHUNK * 2 * 4);
    size_t off_suf    = alloc(NCHUNK * 2 * 4);
    size_t off_cdot   = alloc(NCHUNK * 2 * 4);

    unsigned char* flags = (unsigned char*)(ws + off_flags);
    unsigned* hist   = (unsigned*)(ws + off_hist);
    unsigned* cursor = (unsigned*)(ws + off_cursor);
    unsigned* start  = (unsigned*)(ws + off_start);
    unsigned* keys   = (unsigned*)(ws + off_keys);
    unsigned* vals   = (unsigned*)(ws + off_vals);
    unsigned char* tps = (unsigned char*)(ws + off_tps);
    float* csum  = (float*)(ws + off_csum);
    float* cbase = (float*)(ws + off_cbase);
    float* maxp  = (float*)(ws + off_maxp);
    float* suf   = (float*)(ws + off_suf);
    float* cdot  = (float*)(ws + off_cdot);

    hipMemsetAsync(ws + off_hist, 0, (size_t)NBUCK * 4 * 2, stream);  // hist + cursor

    k_tp<<<BATCH, 256, 0, stream>>>(segments, labels, flags);
    k_hist<<<MTOT / 256, 256, 0, stream>>>(scores, hist);
    k_scan<<<1, 1024, 0, stream>>>(hist, start);
    k_scatter<<<MTOT / 256, 256, 0, stream>>>(scores, flags, start, cursor, keys, vals);
    k_rank<<<NBUCK / 256, 256, 0, stream>>>(hist, start, keys, vals, tps);
    k_chunksum<<<NCHUNK, 256, 0, stream>>>(tps, csum);
    k_basescan<<<1, 32, 0, stream>>>(csum, cbase);
    k_maxp<<<NCHUNK, 256, 0, stream>>>(tps, cbase, maxp);
    k_sufmax<<<1, 32, 0, stream>>>(maxp, suf);
    k_dot<<<NCHUNK, 256, 0, stream>>>(tps, cbase, suf, cdot);
    k_final<<<1, 32, 0, stream>>>(cdot, out);
}
